// GAU_42528766165329
// MI455X (gfx1250) — matrix operand order repaired and checked by reading, not compiled, not run on hardware
//
#include <hip/hip_runtime.h>
#include <hip/hip_bf16.h>

typedef __bf16 bf16;
typedef bf16  bf16x8_t  __attribute__((ext_vector_type(8)));
typedef bf16  bf16x16_t __attribute__((ext_vector_type(16)));
typedef float f32x8_t   __attribute__((ext_vector_type(8)));
typedef unsigned int u32x4_t __attribute__((ext_vector_type(4)));
typedef int i32x4_t __attribute__((ext_vector_type(4)));
typedef int i32x8_t __attribute__((ext_vector_type(8)));

#define BATCH 4
#define SEQ   2048
#define DIM   1024
#define HID   2048
#define QKD   128
#define MROWS (BATCH * SEQ)

__device__ __forceinline__ f32x8_t zero8() {
  f32x8_t z;
#pragma unroll
  for (int i = 0; i < 8; ++i) z[i] = 0.f;
  return z;
}

__device__ __forceinline__ f32x8_t wmma_bf16f32(bf16x16_t a, bf16x16_t b, f32x8_t c) {
  return __builtin_amdgcn_wmma_f32_16x16x32_bf16(false, a, false, b, (short)0, c, false, false);
}

// ---- Tensor Data Mover: 2D bf16 tile -> LDS with row padding (D# per ISA ch.8) ----
__device__ __forceinline__ void tdm_load_2d(unsigned int lds_off, const void* gptr,
                                            unsigned int tile_d0, unsigned int tile_d1,
                                            unsigned int stride0,
                                            unsigned int pad_i, unsigned int pad_a) {
  unsigned long long ga = (unsigned long long)(uintptr_t)gptr;
  u32x4_t g0;
  g0[0] = 1u;                                     // count=1 valid descriptor
  g0[1] = lds_off;                                // lds_addr (bytes)
  g0[2] = (unsigned int)ga;                       // global_addr[31:0]
  g0[3] = (unsigned int)(ga >> 32) | 0x80000000u; // global_addr[56:32] | type=2
  i32x8_t g1;
  g1[0] = (int)((1u << 16) | (1u << 20) | (pad_i << 22) | (pad_a << 25)); // 2B elems, pad
  g1[1] = (int)((tile_d0 & 0xffffu) << 16);
  g1[2] = (int)((tile_d0 >> 16) | ((tile_d1 & 0xffffu) << 16));
  g1[3] = (int)((tile_d1 >> 16) | (tile_d0 << 16));
  g1[4] = (int)(tile_d1 & 0xffffu);
  g1[5] = (int)stride0;
  g1[6] = 0;
  g1[7] = 0;
  i32x4_t z4;
  z4[0] = 0; z4[1] = 0; z4[2] = 0; z4[3] = 0;
#if defined(__clang_major__) && (__clang_major__ >= 23)
  i32x8_t z8;
#pragma unroll
  for (int i = 0; i < 8; ++i) z8[i] = 0;
  __builtin_amdgcn_tensor_load_to_lds(g0, g1, z4, z4, z8, 0);
#else
  __builtin_amdgcn_tensor_load_to_lds(g0, g1, z4, z4, 0);
#endif
}

// A fragment (16x32 bf16): lanes 0-15 row M=lane, k-chunks {0..7,16..23};
// lanes 16-31 same rows, chunks {8..15,24..31}.
__device__ __forceinline__ bf16x16_t lds_load_afrag(const bf16* base, int row0, int k0, int ld, int lane) {
  int row = row0 + (lane & 15);
  int kb  = k0 + ((lane >> 4) << 3);
  const bf16* p = base + row * ld + kb;
  bf16x8_t lo = *(const bf16x8_t*)(p);
  bf16x8_t hi = *(const bf16x8_t*)(p + 16);
  bf16x16_t a;
#pragma unroll
  for (int i = 0; i < 8; ++i) { a[i] = lo[i]; a[i + 8] = hi[i]; }
  return a;
}

// B fragment (32x16 bf16) staged [N][K]: lanes 0-15 column N=lane K={0..7,16..23},
// lanes 16-31 K={8..15,24..31}.
__device__ __forceinline__ bf16x16_t lds_load_bfrag(const bf16* base, int n0, int k0, int ld, int lane) {
  int n  = n0 + (lane & 15);
  int kb = k0 + ((lane >> 4) << 3);
  const bf16* p = base + n * ld + kb;
  bf16x8_t lo = *(const bf16x8_t*)(p);
  bf16x8_t hi = *(const bf16x8_t*)(p + 16);
  bf16x16_t b;
#pragma unroll
  for (int i = 0; i < 8; ++i) { b[i] = lo[i]; b[i + 8] = hi[i]; }
  return b;
}

// ---------------- RMSNorm: x (f32) -> normed (bf16) ----------------
__global__ __launch_bounds__(256) void rmsnorm_kernel(const float* __restrict__ x,
                                                      const float* __restrict__ scale,
                                                      bf16* __restrict__ out) {
  int row = blockIdx.x;
  int tid = threadIdx.x;
  const float* xr = x + (size_t)row * DIM;
  float ss = 0.f;
#pragma unroll
  for (int i = tid; i < DIM; i += 256) { float v = xr[i]; ss += v * v; }
  for (int off = 16; off > 0; off >>= 1) ss += __shfl_down(ss, off, 32);
  __shared__ float red[8];
  int wave = tid >> 5, lane = tid & 31;
  if (lane == 0) red[wave] = ss;
  __syncthreads();
  if (tid == 0) {
    float t = 0.f;
    for (int i = 0; i < 8; ++i) t += red[i];
    red[0] = rsqrtf(t * (1.f / (float)DIM) + 1e-8f);
  }
  __syncthreads();
  float r = red[0];
#pragma unroll
  for (int i = tid; i < DIM; i += 256)
    out[(size_t)row * DIM + i] = (bf16)(scale[i] * xr[i] * r);
}

// -------- weight convert f32 [K][N] -> bf16 [N][K] (transposed) --------
__global__ __launch_bounds__(256) void convert_transpose_kernel(const float* __restrict__ in,
                                                                bf16* __restrict__ out,
                                                                int K, int N) {
  size_t idx = (size_t)blockIdx.x * 256 + threadIdx.x;
  if (idx >= (size_t)K * N) return;
  int n = (int)(idx / K);
  int kk = (int)(idx % K);
  out[(size_t)n * K + kk] = (bf16)in[(size_t)kk * N + n];
}

// -------- v transpose: uv[:, HID:] -> vt [b][h][s] --------
__global__ __launch_bounds__(256) void transpose_v_kernel(const bf16* __restrict__ uv,
                                                          bf16* __restrict__ vt) {
  __shared__ bf16 t[32][33];
  int s0 = blockIdx.x * 32, h0 = blockIdx.y * 32, b = blockIdx.z;
  int tx = threadIdx.x, ty = threadIdx.y;  // 32 x 8
#pragma unroll
  for (int i = 0; i < 32; i += 8)
    t[ty + i][tx] = uv[((size_t)(b * SEQ + s0 + ty + i)) * 4096 + HID + h0 + tx];
  __syncthreads();
#pragma unroll
  for (int i = 0; i < 32; i += 8)
    vt[((size_t)b * HID + h0 + ty + i) * SEQ + s0 + tx] = t[tx][ty + i];
}

// -------- rel-pos bias table: biastab[n] = rel_bias[bucket(n)]*sqrt(QK) --------
__global__ __launch_bounds__(256) void biastab_kernel(const float* __restrict__ rel_bias,
                                                      float* __restrict__ biastab) {
  int n = blockIdx.x * 256 + threadIdx.x;
  if (n >= SEQ) return;
  int bucket = n;
  if (n >= 16) {
    int vb = 16 + (int)(logf((float)n * (1.f / 16.f)) * (16.f / 2.0794415417f)); // /log(8)*16
    bucket = vb > 31 ? 31 : vb;
  }
  biastab[n] = rel_bias[bucket] * 11.313708499f;  // sqrt(128)
}

// ---------------- WMMA GEMM (TDM-staged, double buffered) ----------------
// out = epi(A[M][K] @ Wt[N][K]^T + bias); EPI 0: silu->bf16, EPI 1: f32.
template <int EPI>
__global__ __launch_bounds__(256) void gemm_bf16_kernel(const bf16* __restrict__ A,
                                                        const bf16* __restrict__ Wt,
                                                        const float* __restrict__ bias,
                                                        void* __restrict__ outp,
                                                        int M, int N, int K, int ldo) {
  __shared__ bf16 Alds[2][128 * 40];
  __shared__ bf16 Blds[2][128 * 40];
  int tid = threadIdx.x, lane = tid & 31, wave = tid >> 5;
  int wm = wave & 3, wn = wave >> 2;
  int m0 = blockIdx.y * 128, n0 = blockIdx.x * 128;

  unsigned int aoff[2] = {(unsigned int)(uintptr_t)&Alds[0][0], (unsigned int)(uintptr_t)&Alds[1][0]};
  unsigned int boff[2] = {(unsigned int)(uintptr_t)&Blds[0][0], (unsigned int)(uintptr_t)&Blds[1][0]};

  f32x8_t acc[2][4];
#pragma unroll
  for (int i = 0; i < 2; ++i)
#pragma unroll
    for (int j = 0; j < 4; ++j) acc[i][j] = zero8();

  if (wave == 0) {
    tdm_load_2d(aoff[0], A + (size_t)m0 * K, 32, 128, (unsigned)K, 3, 3);
    tdm_load_2d(boff[0], Wt + (size_t)n0 * K, 32, 128, (unsigned)K, 3, 3);
    __builtin_amdgcn_s_wait_tensorcnt(0);
  }
  __syncthreads();

  int nk = K >> 5;
  for (int ki = 0; ki < nk; ++ki) {
    int cur = ki & 1, nxt = cur ^ 1;
    if (wave == 0 && ki + 1 < nk) {
      tdm_load_2d(aoff[nxt], A + (size_t)m0 * K + (size_t)(ki + 1) * 32, 32, 128, (unsigned)K, 3, 3);
      tdm_load_2d(boff[nxt], Wt + (size_t)n0 * K + (size_t)(ki + 1) * 32, 32, 128, (unsigned)K, 3, 3);
    }
    bf16x16_t afrag[2], bfrag[4];
#pragma unroll
    for (int mi = 0; mi < 2; ++mi) afrag[mi] = lds_load_afrag(&Alds[cur][0], wm * 32 + mi * 16, 0, 40, lane);
#pragma unroll
    for (int ni = 0; ni < 4; ++ni) bfrag[ni] = lds_load_bfrag(&Blds[cur][0], wn * 64 + ni * 16, 0, 40, lane);
#pragma unroll
    for (int mi = 0; mi < 2; ++mi)
#pragma unroll
      for (int ni = 0; ni < 4; ++ni) acc[mi][ni] = wmma_bf16f32(afrag[mi], bfrag[ni], acc[mi][ni]);
    if (wave == 0) __builtin_amdgcn_s_wait_tensorcnt(0);
    __syncthreads();
  }

  int lhalf = lane >> 4, lcol = lane & 15;
#pragma unroll
  for (int mi = 0; mi < 2; ++mi)
#pragma unroll
    for (int ni = 0; ni < 4; ++ni)
#pragma unroll
      for (int r = 0; r < 8; ++r) {
        int grow = m0 + wm * 32 + mi * 16 + lhalf * 8 + r;
        int gcol = n0 + wn * 64 + ni * 16 + lcol;
        float v = acc[mi][ni][r] + bias[gcol];
        if (EPI == 0) {
          float s = v / (1.f + expf(-v));  // silu
          ((bf16*)outp)[(size_t)grow * ldo + gcol] = (bf16)s;
        } else {
          ((float*)outp)[(size_t)grow * ldo + gcol] = v;
        }
      }
}

// ---------------- RoPE + xPos ----------------
__global__ __launch_bounds__(128) void rope_kernel(const bf16* __restrict__ qk,
                                                   const float* __restrict__ gamma,
                                                   const float* __restrict__ beta,
                                                   bf16* __restrict__ qo,
                                                   bf16* __restrict__ ko) {
  int row = blockIdx.x;
  int j = threadIdx.x;
  int s = row & (SEQ - 1);
  float xj = (float)qk[(size_t)row * QKD + j];
  float xp = (float)qk[(size_t)row * QKD + (j ^ 1)];
  float qj  = xj * gamma[j] + beta[j];
  float qpp = xp * gamma[j ^ 1] + beta[j ^ 1];
  float kj  = xj * gamma[QKD + j] + beta[QKD + j];
  float kpp = xp * gamma[QKD + (j ^ 1)] + beta[QKD + (j ^ 1)];
  float rq = (j & 1) ? qpp : -qpp;
  float rk = (j & 1) ? kpp : -kpp;
  int i = j >> 1;
  float invf = powf(10000.f, -(2.f * (float)i) / (float)QKD);
  float fr = (float)s * invf;
  float c = cosf(fr), sn = sinf(fr);
  int ix = j & 63;
  float sv = (2.f * (float)ix + 0.4f * (float)QKD) / (1.4f * (float)QKD);
  float pw = ((float)s - (float)(QKD / 2)) / 512.f;
  float xps = powf(sv, pw);
  qo[(size_t)row * QKD + j] = (bf16)((qj * c + rq * sn) * xps);
  ko[(size_t)row * QKD + j] = (bf16)((kj * c + rk * sn) / xps);
}

// ---------------- sim: P = laplacian(q @ k^T + bias) with causal mask ----------------
// one causal 128x128 tile per block; q frags in registers, k tile via TDM.
__global__ __launch_bounds__(256) void sim_kernel(const bf16* __restrict__ q,
                                                  const bf16* __restrict__ k,
                                                  const float* __restrict__ biastab,
                                                  bf16* __restrict__ P) {
  int kx = blockIdx.x, qy = blockIdx.y, b = blockIdx.z;
  if (kx > qy) return;  // uniform: strictly-upper blocks never read by pv
  __shared__ bf16 k_lds[128 * 136];
  __shared__ float bt[256];
  int tid = threadIdx.x, lane = tid & 31, wave = tid >> 5;
  int q0 = qy * 128, k0 = kx * 128;
  const size_t rowbase = (size_t)b * SEQ;

  {  // bias band: this tile only spans distances (q0-k0) + [-127,127]
    int n = q0 - k0 - 127 + tid;
    bt[tid] = (n >= 0 && n < SEQ) ? biastab[n] : 0.f;
  }
  if (wave == 0) {
    tdm_load_2d((unsigned int)(uintptr_t)&k_lds[0], k + (rowbase + k0) * (size_t)QKD,
                128, 128, QKD, 5, 3);
    __builtin_amdgcn_s_wait_tensorcnt(0);
  }

  int wm = wave & 3, wn = wave >> 2;
  bf16x16_t qfrag[2][4];
  {
    int row = lane & 15;
#pragma unroll
    for (int mi = 0; mi < 2; ++mi) {
      const bf16* qb = q + (rowbase + q0 + wm * 32 + mi * 16 + row) * (size_t)QKD;
#pragma unroll
      for (int ks = 0; ks < 4; ++ks) {
        int kc = ks * 32 + ((lane >> 4) << 3);
        bf16x8_t lo = *(const bf16x8_t*)(qb + kc);
        bf16x8_t hi = *(const bf16x8_t*)(qb + kc + 16);
#pragma unroll
        for (int i = 0; i < 8; ++i) { qfrag[mi][ks][i] = lo[i]; qfrag[mi][ks][i + 8] = hi[i]; }
      }
    }
  }
  __syncthreads();

  f32x8_t acc[2][4];
#pragma unroll
  for (int i = 0; i < 2; ++i)
#pragma unroll
    for (int j = 0; j < 4; ++j) acc[i][j] = zero8();
#pragma unroll
  for (int ks = 0; ks < 4; ++ks) {
    bf16x16_t bfrag[4];
#pragma unroll
    for (int ni = 0; ni < 4; ++ni) bfrag[ni] = lds_load_bfrag(k_lds, wn * 64 + ni * 16, ks * 32, 136, lane);
#pragma unroll
    for (int mi = 0; mi < 2; ++mi)
#pragma unroll
      for (int ni = 0; ni < 4; ++ni) acc[mi][ni] = wmma_bf16f32(qfrag[mi][ks], bfrag[ni], acc[mi][ni]);
  }

  const float MU   = 0.70710678118f;                       // sqrt(0.5)
  const float INVD = 1.f / (0.88622692545f * 45.254834f);  // 1/(sqrt(pi/4)*sqrt(S))
  int lhalf = lane >> 4, lcol = lane & 15;
#pragma unroll
  for (int mi = 0; mi < 2; ++mi)
#pragma unroll
    for (int ni = 0; ni < 4; ++ni)
#pragma unroll
      for (int r = 0; r < 8; ++r) {
        int lq = wm * 32 + mi * 16 + lhalf * 8 + r;
        int lk = wn * 64 + ni * 16 + lcol;
        int sq = q0 + lq, sk = k0 + lk;
        float simv = acc[mi][ni][r] + bt[lq - lk + 127];
        float e = 0.5f * (1.f + erff((simv - MU) * INVD));
        float val = (sk <= sq) ? e : 0.f;
        P[(rowbase + sq) * (size_t)SEQ + sk] = (bf16)val;
      }
}

// ---------------- pv: gated = u * (P @ v), TDM double buffered, causal K ----------------
__global__ __launch_bounds__(256) void pv_kernel(const bf16* __restrict__ P,
                                                 const bf16* __restrict__ vt,
                                                 const bf16* __restrict__ uv,
                                                 bf16* __restrict__ gated) {
  __shared__ bf16 Alds[2][128 * 40];
  __shared__ bf16 Blds[2][128 * 40];
  int tid = threadIdx.x, lane = tid & 31, wave = tid >> 5;
  int wm = wave & 3, wn = wave >> 2;
  int n0 = blockIdx.x * 128;          // h
  int m0 = blockIdx.y * 128;          // q
  int b  = blockIdx.z;
  const bf16* A = P  + (size_t)b * SEQ * SEQ;   // [S][S]
  const bf16* W = vt + (size_t)b * HID * SEQ;   // [HID][S]

  unsigned int aoff[2] = {(unsigned int)(uintptr_t)&Alds[0][0], (unsigned int)(uintptr_t)&Alds[1][0]};
  unsigned int boff[2] = {(unsigned int)(uintptr_t)&Blds[0][0], (unsigned int)(uintptr_t)&Blds[1][0]};

  f32x8_t acc[2][4];
#pragma unroll
  for (int i = 0; i < 2; ++i)
#pragma unroll
    for (int j = 0; j < 4; ++j) acc[i][j] = zero8();

  if (wave == 0) {
    tdm_load_2d(aoff[0], A + (size_t)m0 * SEQ, 32, 128, SEQ, 3, 3);
    tdm_load_2d(boff[0], W + (size_t)n0 * SEQ, 32, 128, SEQ, 3, 3);
    __builtin_amdgcn_s_wait_tensorcnt(0);
  }
  __syncthreads();

  int nk = (blockIdx.y + 1) * 4;  // causal: keys up to (m0+128)
  for (int ki = 0; ki < nk; ++ki) {
    int cur = ki & 1, nxt = cur ^ 1;
    if (wave == 0 && ki + 1 < nk) {
      tdm_load_2d(aoff[nxt], A + (size_t)m0 * SEQ + (size_t)(ki + 1) * 32, 32, 128, SEQ, 3, 3);
      tdm_load_2d(boff[nxt], W + (size_t)n0 * SEQ + (size_t)(ki + 1) * 32, 32, 128, SEQ, 3, 3);
    }
    bf16x16_t afrag[2], bfrag[4];
#pragma unroll
    for (int mi = 0; mi < 2; ++mi) afrag[mi] = lds_load_afrag(&Alds[cur][0], wm * 32 + mi * 16, 0, 40, lane);
#pragma unroll
    for (int ni = 0; ni < 4; ++ni) bfrag[ni] = lds_load_bfrag(&Blds[cur][0], wn * 64 + ni * 16, 0, 40, lane);
#pragma unroll
    for (int mi = 0; mi < 2; ++mi)
#pragma unroll
      for (int ni = 0; ni < 4; ++ni) acc[mi][ni] = wmma_bf16f32(afrag[mi], bfrag[ni], acc[mi][ni]);
    if (wave == 0) __builtin_amdgcn_s_wait_tensorcnt(0);
    __syncthreads();
  }

  int lhalf = lane >> 4, lcol = lane & 15;
#pragma unroll
  for (int mi = 0; mi < 2; ++mi)
#pragma unroll
    for (int ni = 0; ni < 4; ++ni)
#pragma unroll
      for (int r = 0; r < 8; ++r) {
        int row = m0 + wm * 32 + mi * 16 + lhalf * 8 + r;
        int col = n0 + wn * 64 + ni * 16 + lcol;
        size_t grow = (size_t)b * SEQ + row;
        float uval = (float)uv[grow * 4096 + col];
        gated[grow * (size_t)HID + col] = (bf16)(acc[mi][ni][r] * uval);
      }
}

extern "C" void kernel_launch(void* const* d_in, const int* in_sizes, int n_in,
                              void* d_out, int out_size, void* d_ws, size_t ws_size,
                              hipStream_t stream) {
  const float* x        = (const float*)d_in[0];
  const float* rscale   = (const float*)d_in[1];
  const float* w_hidden = (const float*)d_in[2];
  const float* b_hidden = (const float*)d_in[3];
  const float* w_qk     = (const float*)d_in[4];
  const float* b_qk     = (const float*)d_in[5];
  const float* gamma    = (const float*)d_in[6];
  const float* beta     = (const float*)d_in[7];
  const float* rel_bias = (const float*)d_in[8];
  const float* w_out    = (const float*)d_in[9];
  const float* b_out    = (const float*)d_in[10];
  float* out = (float*)d_out;

  char* ws = (char*)d_ws;
  size_t off = 0;
  auto take = [&](size_t bytes) -> char* {
    char* p = ws + off;
    off += (bytes + 255) & ~(size_t)255;
    return p;
  };
  bf16* normed  = (bf16*)take((size_t)MROWS * DIM * 2);
  bf16* wh_t    = (bf16*)take((size_t)2 * HID * DIM * 2);
  bf16* wqk_t   = (bf16*)take((size_t)QKD * DIM * 2);
  bf16* wout_t  = (bf16*)take((size_t)DIM * HID * 2);
  bf16* uvbuf   = (bf16*)take((size_t)MROWS * 4096 * 2);
  bf16* qksilu  = (bf16*)take((size_t)MROWS * QKD * 2);
  bf16* qbuf    = (bf16*)take((size_t)MROWS * QKD * 2);
  bf16* kbuf    = (bf16*)take((size_t)MROWS * QKD * 2);
  bf16* vtbuf   = (bf16*)take((size_t)BATCH * HID * SEQ * 2);
  bf16* gated   = (bf16*)take((size_t)MROWS * HID * 2);
  bf16* pbuf    = (bf16*)take((size_t)BATCH * SEQ * SEQ * 2);
  float* btab   = (float*)take((size_t)SEQ * 4);

  rmsnorm_kernel<<<MROWS, 256, 0, stream>>>(x, rscale, normed);
  convert_transpose_kernel<<<(DIM * 2 * HID) / 256, 256, 0, stream>>>(w_hidden, wh_t, DIM, 2 * HID);
  convert_transpose_kernel<<<(DIM * QKD) / 256, 256, 0, stream>>>(w_qk, wqk_t, DIM, QKD);
  convert_transpose_kernel<<<(HID * DIM) / 256, 256, 0, stream>>>(w_out, wout_t, HID, DIM);
  biastab_kernel<<<SEQ / 256, 256, 0, stream>>>(rel_bias, btab);

  gemm_bf16_kernel<0><<<dim3(4096 / 128, MROWS / 128), 256, 0, stream>>>(
      normed, wh_t, b_hidden, uvbuf, MROWS, 4096, DIM, 4096);
  gemm_bf16_kernel<0><<<dim3(QKD / 128, MROWS / 128), 256, 0, stream>>>(
      normed, wqk_t, b_qk, qksilu, MROWS, QKD, DIM, QKD);
  rope_kernel<<<MROWS, QKD, 0, stream>>>(qksilu, gamma, beta, qbuf, kbuf);
  transpose_v_kernel<<<dim3(SEQ / 32, HID / 32, BATCH), dim3(32, 8), 0, stream>>>(uvbuf, vtbuf);

  sim_kernel<<<dim3(SEQ / 128, SEQ / 128, BATCH), 256, 0, stream>>>(qbuf, kbuf, btab, pbuf);
  pv_kernel<<<dim3(HID / 128, SEQ / 128, BATCH), 256, 0, stream>>>(pbuf, vtbuf, uvbuf, gated);

  gemm_bf16_kernel<1><<<dim3(DIM / 128, MROWS / 128), 256, 0, stream>>>(
      gated, wout_t, b_out, out, MROWS, DIM, HID, DIM);
}
